// MultiHeadAttention_35416300323576
// MI455X (gfx1250) — compile-verified, matrix operand order checked by reading, not run on hardware
//
#include <hip/hip_runtime.h>

// ---------------------------------------------------------------------------
// Multi-head attention forward for MI455X (gfx1250), wave32 + WMMA f16 + TDM.
//   d_out = [ out (4*2048*1024 f32) | attn (4*16*2048*2048 f32) ]
//   d_ws  = [ Qh f16 16MB | Kh f16 16MB | Vh f16 16MB | ctx f16 16MB |
//             rowmax f32 512KB | rowrinv f32 512KB ]  (~65 MB used)
// ---------------------------------------------------------------------------

typedef __attribute__((ext_vector_type(16))) _Float16 v16h;
typedef __attribute__((ext_vector_type(8)))  float    v8f;
typedef __attribute__((ext_vector_type(4)))  unsigned int v4u;
typedef __attribute__((ext_vector_type(8)))  int      v8i;
typedef __attribute__((ext_vector_type(4)))  int      v4i;

#define SEQ   2048
#define DMODEL 1024
#define NH    16
#define DKH   64
#define BSZ   4
#define BHCNT (BSZ*NH)          // 64
#define MTOT  (BSZ*SEQ)         // 8192
#define SCALE 0.125f            // 1/sqrt(64)

#define AP 40                   // LDS pitch (halves) for 32-wide K tiles
#define QP 72                   // LDS pitch (halves) for 64-wide K tiles

struct __align__(8) H4 { _Float16 x, y, z, w; };
union F16x16 { v16h v; float4 f4[2]; };
union F16x8  { float4 f4; _Float16 h[8]; };

// Load one 16x32 f16 WMMA fragment (A layout; also B layout when the tile is
// stored transposed as lds[n][k]).  Lane L: row = L&15, k-half = L>>4.
__device__ inline v16h ld_frag(const _Float16* base, int pitch, int lane) {
  const int r  = lane & 15;
  const int kh = (lane >> 4) << 3;
  const _Float16* p = base + r * pitch + kh;
  F16x16 u;
  u.f4[0] = *(const float4*)(p);
  u.f4[1] = *(const float4*)(p + 16);
  return u.v;
}

__device__ inline v8f wmma16(v16h a, v16h b, v8f c) {
  return __builtin_amdgcn_wmma_f32_16x16x32_f16(false, a, false, b,
                                                (short)0, c, false, false);
}

__device__ inline v8f vzero8() {
  v8f z;
#pragma unroll
  for (int i = 0; i < 8; ++i) z[i] = 0.f;
  return z;
}

// ---------------------------------------------------------------------------
// TDM: load a 2-D f16 tile (rows x 64 halves, global row stride 64 halves)
// into LDS with 8-half row padding (LDS pitch = 72 halves = QP).
//   row bytes = 128 B = 32 DWORDs  -> pad_interval = 4  (2^(4+1) = 32 DWORDs)
//   pad       =  16 B =  4 DWORDs  -> pad_amount   = 3  (3+1   =  4 DWORDs)
// D# group0: count=1 | lds_addr | global_addr[56:0] | type=2.
// D# group1: data_size=1(2B), pad_enable, tensor 64 x rows, tile 64 x rows,
//            dim0 stride 64.  Groups 2/3 zero (2-D tensor).
// This toolchain's builtin is the 6-arg form:
//   (v4u g0, v8i g1, v4i g2, v4i g3, v8i, i32 cpol)
// ---------------------------------------------------------------------------
__device__ inline void tdm_load_tile_f16(unsigned lds_addr,
                                         const _Float16* gptr, int rows) {
  unsigned long long ga = (unsigned long long)(uintptr_t)gptr;
  v4u g0;
  g0[0] = 1u;                                         // count = 1 valid D#
  g0[1] = lds_addr;                                   // LDS byte address
  g0[2] = (unsigned)(ga & 0xffffffffu);               // global_addr[31:0]
  g0[3] = (unsigned)((ga >> 32) & 0x1ffffffu) | (2u << 30);  // [56:32]|type=2
  v8i g1;
  g1[0] = (int)((1u << 16) | (1u << 20) | (4u << 22) | (3u << 25));
  g1[1] = (int)(64u << 16);                 // tensor_dim0 = 64 (bits 79:48)
  g1[2] = (int)(((unsigned)rows) << 16);    // tensor_dim1 = rows (bits 111:80)
  g1[3] = (int)(64u << 16);                 // tile_dim0 = 64 (bits 127:112)
  g1[4] = rows;                             // tile_dim1 = rows, tile_dim2 = 0
  g1[5] = 64;                               // tensor_dim0_stride = 64
  g1[6] = 0;
  g1[7] = 0;
  v4i z4 = {0, 0, 0, 0};
  v8i z8 = {0, 0, 0, 0, 0, 0, 0, 0};
  __builtin_amdgcn_tensor_load_to_lds(g0, g1, z4, z4, z8, 0);
}

// ---------------------------------------------------------------------------
// Kernel 1: projection  Out_f16[head-split] = X(f32) @ W(f32) + bias
// grid (DMODEL/128, MTOT/128), block 256.  Wave tile 32x64 (2x4 16x16 tiles).
// ---------------------------------------------------------------------------
__global__ __launch_bounds__(256) void mha_proj_gemm(
    const float* __restrict__ X, const float* __restrict__ W,
    const float* __restrict__ bias, _Float16* __restrict__ Out) {
  __shared__ _Float16 Alds[128 * AP];
  __shared__ _Float16 Blds[128 * AP];
  const int tid  = threadIdx.x;
  const int lane = tid & 31, wid = tid >> 5;
  const int wr = wid & 3, wc = wid >> 2;          // rows 32*wr, cols 64*wc
  const int m0 = blockIdx.y * 128, n0 = blockIdx.x * 128;

  v8f acc[2][4];
#pragma unroll
  for (int mt = 0; mt < 2; ++mt)
#pragma unroll
    for (int nt = 0; nt < 4; ++nt) acc[mt][nt] = vzero8();

  for (int k0 = 0; k0 < DMODEL; k0 += 32) {
    // branchless next-tile prefetch offset (uniform s_cselect, always valid)
    const int koff = (k0 + 32 < DMODEL) ? 32 : 0;
    // stage A: 128x32 f32 -> f16, row-major
#pragma unroll
    for (int i = 0; i < 4; ++i) {
      int t = tid + i * 256;
      int row = t >> 3, c = (t & 7) * 4;
      const float* gp = X + (size_t)(m0 + row) * DMODEL + k0 + c;
      float4 v = *(const float4*)(gp);
      __builtin_prefetch(gp + koff, 0, 3);
      H4 h4 = { (_Float16)v.x, (_Float16)v.y, (_Float16)v.z, (_Float16)v.w };
      *(H4*)(&Alds[row * AP + c]) = h4;
    }
    // stage B transposed: lds[n][k] = W[k0+k][n0+n]
#pragma unroll
    for (int i = 0; i < 4; ++i) {
      int t = tid + i * 256;
      int kr = t >> 5, c = (t & 31) * 4;
      const float* gp = W + (size_t)(k0 + kr) * DMODEL + n0 + c;
      float4 v = *(const float4*)(gp);
      __builtin_prefetch(gp + (size_t)koff * DMODEL, 0, 3);
      Blds[(c + 0) * AP + kr] = (_Float16)v.x;
      Blds[(c + 1) * AP + kr] = (_Float16)v.y;
      Blds[(c + 2) * AP + kr] = (_Float16)v.z;
      Blds[(c + 3) * AP + kr] = (_Float16)v.w;
    }
    __syncthreads();
    v16h a[2], b[4];
#pragma unroll
    for (int mt = 0; mt < 2; ++mt)
      a[mt] = ld_frag(&Alds[(32 * wr + 16 * mt) * AP], AP, lane);
#pragma unroll
    for (int nt = 0; nt < 4; ++nt)
      b[nt] = ld_frag(&Blds[(64 * wc + 16 * nt) * AP], AP, lane);
#pragma unroll
    for (int mt = 0; mt < 2; ++mt)
#pragma unroll
      for (int nt = 0; nt < 4; ++nt)
        acc[mt][nt] = wmma16(a[mt], b[nt], acc[mt][nt]);
    __syncthreads();
  }

  // epilogue: + bias, head-split f16 store  [bh][s][dk]
#pragma unroll
  for (int mt = 0; mt < 2; ++mt)
#pragma unroll
    for (int nt = 0; nt < 4; ++nt) {
      int ng = n0 + 64 * wc + 16 * nt + (lane & 15);
      float bv = bias[ng];
      int h = ng >> 6, d = ng & 63;
#pragma unroll
      for (int r = 0; r < 8; ++r) {
        int mg = m0 + 32 * wr + 16 * mt + (lane >> 4) * 8 + r;
        int bb = mg >> 11, s = mg & 2047;
        Out[((size_t)(bb * NH + h) * SEQ + s) * DKH + d] =
            (_Float16)(acc[mt][nt][r] + bv);
      }
    }
}

// ---------------------------------------------------------------------------
// Kernel 2: raw scores  attn[bh][q][k] = SCALE * Qh[bh] @ Kh[bh]^T   (f32)
// grid (SEQ/128, SEQ/128, BHCNT), block 256.  Q/K tiles staged by the TDM.
// ---------------------------------------------------------------------------
__global__ __launch_bounds__(256) void mha_score_gemm(
    const _Float16* __restrict__ Qh, const _Float16* __restrict__ Kh,
    float* __restrict__ attn) {
  __shared__ _Float16 Ql[128 * QP];
  __shared__ _Float16 Kl[128 * QP];
  const int tid  = threadIdx.x;
  const int lane = tid & 31, wid = tid >> 5;
  const int wr = wid & 3, wc = wid >> 2;
  const int bh = blockIdx.z;
  const int m0 = blockIdx.y * 128, n0 = blockIdx.x * 128;

  // Tensor Data Mover: wave 0 issues both tile DMAs (padding to pitch QP
  // applied by the TDM), waits TENSORcnt==0, then the whole block syncs.
  if (wid == 0) {
    tdm_load_tile_f16((unsigned)(uintptr_t)&Ql[0],
                      Qh + ((size_t)bh * SEQ + m0) * DKH, 128);
    tdm_load_tile_f16((unsigned)(uintptr_t)&Kl[0],
                      Kh + ((size_t)bh * SEQ + n0) * DKH, 128);
    __builtin_amdgcn_s_wait_tensorcnt(0);
  }
  __syncthreads();

  v8f acc[2][4];
#pragma unroll
  for (int mt = 0; mt < 2; ++mt)
#pragma unroll
    for (int nt = 0; nt < 4; ++nt) acc[mt][nt] = vzero8();

#pragma unroll
  for (int kk = 0; kk < DKH; kk += 32) {
    v16h a[2], b[4];
#pragma unroll
    for (int mt = 0; mt < 2; ++mt)
      a[mt] = ld_frag(&Ql[(32 * wr + 16 * mt) * QP + kk], QP, lane);
#pragma unroll
    for (int nt = 0; nt < 4; ++nt)
      b[nt] = ld_frag(&Kl[(64 * wc + 16 * nt) * QP + kk], QP, lane);
#pragma unroll
    for (int mt = 0; mt < 2; ++mt)
#pragma unroll
      for (int nt = 0; nt < 4; ++nt)
        acc[mt][nt] = wmma16(a[mt], b[nt], acc[mt][nt]);
  }

#pragma unroll
  for (int mt = 0; mt < 2; ++mt)
#pragma unroll
    for (int nt = 0; nt < 4; ++nt) {
      int ng = n0 + 64 * wc + 16 * nt + (lane & 15);
#pragma unroll
      for (int r = 0; r < 8; ++r) {
        int mg = m0 + 32 * wr + 16 * mt + (lane >> 4) * 8 + r;
        attn[((size_t)bh * SEQ + mg) * SEQ + ng] = acc[mt][nt][r] * SCALE;
      }
    }
}

// ---------------------------------------------------------------------------
// Kernel 3: per-row max and 1/sum(exp(s-max)).  One 256-thread block per row.
// ---------------------------------------------------------------------------
__global__ __launch_bounds__(256) void mha_row_stats(
    const float* __restrict__ attn, float* __restrict__ rowmax,
    float* __restrict__ rowrinv) {
  __shared__ float red[256];
  const int tid = threadIdx.x;
  const size_t row = blockIdx.x;
  const float* p = attn + row * SEQ;
  float v[8];
  float mx = -3.4e38f;
#pragma unroll
  for (int i = 0; i < 8; ++i) { v[i] = p[tid + i * 256]; mx = fmaxf(mx, v[i]); }
  red[tid] = mx;
  __syncthreads();
  for (int s = 128; s > 0; s >>= 1) {
    if (tid < s) red[tid] = fmaxf(red[tid], red[tid + s]);
    __syncthreads();
  }
  mx = red[0];
  __syncthreads();
  float sum = 0.f;
#pragma unroll
  for (int i = 0; i < 8; ++i) sum += __expf(v[i] - mx);
  red[tid] = sum;
  __syncthreads();
  for (int s = 128; s > 0; s >>= 1) {
    if (tid < s) red[tid] += red[tid + s];
    __syncthreads();
  }
  if (tid == 0) { rowmax[row] = mx; rowrinv[row] = 1.0f / red[0]; }
}

// ---------------------------------------------------------------------------
// Kernel 4: normalize attn in place and context = P @ V  (ctx f16).
// grid (SEQ/128, BHCNT), block 256.  Wave tile 32x32 (2x2).
// ---------------------------------------------------------------------------
__global__ __launch_bounds__(256) void mha_pv_gemm(
    float* __restrict__ attn, const float* __restrict__ rowmax,
    const float* __restrict__ rowrinv, const _Float16* __restrict__ Vh,
    _Float16* __restrict__ ctx) {
  __shared__ _Float16 Pl[128 * AP];   // 128 q-rows x 32 keys
  __shared__ _Float16 Vl[64 * AP];    // 64 dk-rows x 32 keys (transposed V)
  const int tid  = threadIdx.x;
  const int lane = tid & 31, wid = tid >> 5;
  const int wr = wid & 3, wc = wid >> 2;          // rows 32*wr, cols 32*wc
  const int bh = blockIdx.y;
  const int m0 = blockIdx.x * 128;
  float* abase = attn + ((size_t)bh * SEQ + m0) * SEQ;

  v8f acc[2][2];
#pragma unroll
  for (int mt = 0; mt < 2; ++mt)
#pragma unroll
    for (int nt = 0; nt < 2; ++nt) acc[mt][nt] = vzero8();

  for (int k0 = 0; k0 < SEQ; k0 += 32) {
    const int koff = (k0 + 32 < SEQ) ? 32 : 0;
    // stage P tile: softmax-normalize, write back to attn, f16 copy to LDS
#pragma unroll
    for (int i = 0; i < 4; ++i) {
      int t = tid + i * 256;
      int row = t >> 3, c = (t & 7) * 4;
      float* sp = abase + (size_t)row * SEQ + k0 + c;
      float4 sv = *(const float4*)(sp);
      __builtin_prefetch(sp + koff, 0, 3);
      float mxv = rowmax[(size_t)bh * SEQ + m0 + row];
      float rv  = rowrinv[(size_t)bh * SEQ + m0 + row];
      float4 pv;
      pv.x = __expf(sv.x - mxv) * rv;
      pv.y = __expf(sv.y - mxv) * rv;
      pv.z = __expf(sv.z - mxv) * rv;
      pv.w = __expf(sv.w - mxv) * rv;
      *(float4*)(sp) = pv;                          // final attn output
      H4 h4 = { (_Float16)pv.x, (_Float16)pv.y, (_Float16)pv.z, (_Float16)pv.w };
      *(H4*)(&Pl[row * AP + c]) = h4;
    }
    // stage V transposed: Vl[d][key] = Vh[bh][k0+key][d]
    {
      int key = tid >> 3, c = (tid & 7) * 8;
      F16x8 u;
      u.f4 = *(const float4*)(Vh + ((size_t)bh * SEQ + k0 + key) * DKH + c);
#pragma unroll
      for (int j = 0; j < 8; ++j) Vl[(c + j) * AP + key] = u.h[j];
    }
    __syncthreads();
    v16h a[2], b[2];
#pragma unroll
    for (int mt = 0; mt < 2; ++mt)
      a[mt] = ld_frag(&Pl[(32 * wr + 16 * mt) * AP], AP, lane);
#pragma unroll
    for (int nt = 0; nt < 2; ++nt)
      b[nt] = ld_frag(&Vl[(32 * wc + 16 * nt) * AP], AP, lane);
#pragma unroll
    for (int mt = 0; mt < 2; ++mt)
#pragma unroll
      for (int nt = 0; nt < 2; ++nt)
        acc[mt][nt] = wmma16(a[mt], b[nt], acc[mt][nt]);
    __syncthreads();
  }

  // epilogue: ctx[b*SEQ+s][h*64+d] f16 (merged-head layout for out proj)
  const int bb = bh >> 4, h = bh & 15;
#pragma unroll
  for (int mt = 0; mt < 2; ++mt)
#pragma unroll
    for (int nt = 0; nt < 2; ++nt) {
      int d = 32 * wc + 16 * nt + (lane & 15);
#pragma unroll
      for (int r = 0; r < 8; ++r) {
        int s = m0 + 32 * wr + 16 * mt + (lane >> 4) * 8 + r;
        ctx[((size_t)(bb * SEQ + s)) * DMODEL + h * DKH + d] =
            (_Float16)acc[mt][nt][r];
      }
    }
}

// ---------------------------------------------------------------------------
// Kernel 5: out = ctx(f16) @ Wo(f32) + bo   (f32, plain [m][n])
// grid (DMODEL/128, MTOT/128), block 256.
// ---------------------------------------------------------------------------
__global__ __launch_bounds__(256) void mha_out_gemm(
    const _Float16* __restrict__ A, const float* __restrict__ W,
    const float* __restrict__ bias, float* __restrict__ Out) {
  __shared__ _Float16 Alds[128 * AP];
  __shared__ _Float16 Blds[128 * AP];
  const int tid  = threadIdx.x;
  const int lane = tid & 31, wid = tid >> 5;
  const int wr = wid & 3, wc = wid >> 2;
  const int m0 = blockIdx.y * 128, n0 = blockIdx.x * 128;

  v8f acc[2][4];
#pragma unroll
  for (int mt = 0; mt < 2; ++mt)
#pragma unroll
    for (int nt = 0; nt < 4; ++nt) acc[mt][nt] = vzero8();

  for (int k0 = 0; k0 < DMODEL; k0 += 32) {
    const int koff = (k0 + 32 < DMODEL) ? 32 : 0;
#pragma unroll
    for (int i = 0; i < 2; ++i) {                   // A already f16
      int t = tid + i * 256;
      int row = t >> 2, c = (t & 3) * 8;
      const _Float16* gp = A + (size_t)(m0 + row) * DMODEL + k0 + c;
      float4 v = *(const float4*)(gp);
      __builtin_prefetch(gp + koff, 0, 3);
      *(float4*)(&Alds[row * AP + c]) = v;
    }
#pragma unroll
    for (int i = 0; i < 4; ++i) {                   // B transposed f32->f16
      int t = tid + i * 256;
      int kr = t >> 5, c = (t & 31) * 4;
      const float* gp = W + (size_t)(k0 + kr) * DMODEL + n0 + c;
      float4 v = *(const float4*)(gp);
      __builtin_prefetch(gp + (size_t)koff * DMODEL, 0, 3);
      Blds[(c + 0) * AP + kr] = (_Float16)v.x;
      Blds[(c + 1) * AP + kr] = (_Float16)v.y;
      Blds[(c + 2) * AP + kr] = (_Float16)v.z;
      Blds[(c + 3) * AP + kr] = (_Float16)v.w;
    }
    __syncthreads();
    v16h a[2], b[4];
#pragma unroll
    for (int mt = 0; mt < 2; ++mt)
      a[mt] = ld_frag(&Alds[(32 * wr + 16 * mt) * AP], AP, lane);
#pragma unroll
    for (int nt = 0; nt < 4; ++nt)
      b[nt] = ld_frag(&Blds[(64 * wc + 16 * nt) * AP], AP, lane);
#pragma unroll
    for (int mt = 0; mt < 2; ++mt)
#pragma unroll
      for (int nt = 0; nt < 4; ++nt)
        acc[mt][nt] = wmma16(a[mt], b[nt], acc[mt][nt]);
    __syncthreads();
  }

#pragma unroll
  for (int mt = 0; mt < 2; ++mt)
#pragma unroll
    for (int nt = 0; nt < 4; ++nt) {
      int ng = n0 + 64 * wc + 16 * nt + (lane & 15);
      float bv = bias[ng];
#pragma unroll
      for (int r = 0; r < 8; ++r) {
        int mg = m0 + 32 * wr + 16 * mt + (lane >> 4) * 8 + r;
        Out[(size_t)mg * DMODEL + ng] = acc[mt][nt][r] + bv;
      }
    }
}

// ---------------------------------------------------------------------------
extern "C" void kernel_launch(void* const* d_in, const int* in_sizes, int n_in,
                              void* d_out, int out_size, void* d_ws,
                              size_t ws_size, hipStream_t stream) {
  const float* query = (const float*)d_in[0];
  const float* key_  = (const float*)d_in[1];
  const float* value = (const float*)d_in[2];
  const float* Wq = (const float*)d_in[3];
  const float* bq = (const float*)d_in[4];
  const float* Wk = (const float*)d_in[5];
  const float* bk = (const float*)d_in[6];
  const float* Wv = (const float*)d_in[7];
  const float* bv = (const float*)d_in[8];
  const float* Wo = (const float*)d_in[9];
  const float* bo = (const float*)d_in[10];

  float* out  = (float*)d_out;
  float* attn = out + (size_t)BSZ * SEQ * DMODEL;   // 8388608 floats in

  char* w = (char*)d_ws;
  const size_t HBYTES = (size_t)MTOT * DMODEL * sizeof(_Float16);  // 16 MB
  _Float16* Qh  = (_Float16*)(w);
  _Float16* Kh  = (_Float16*)(w + HBYTES);
  _Float16* Vh  = (_Float16*)(w + 2 * HBYTES);
  _Float16* ctx = (_Float16*)(w + 3 * HBYTES);
  float* rowmax  = (float*)(w + 4 * HBYTES);
  float* rowrinv = rowmax + (size_t)BHCNT * SEQ;

  dim3 blk(256);
  dim3 gProj(DMODEL / 128, MTOT / 128);             // (8, 64)
  mha_proj_gemm<<<gProj, blk, 0, stream>>>(query, Wq, bq, Qh);
  mha_proj_gemm<<<gProj, blk, 0, stream>>>(key_,  Wk, bk, Kh);
  mha_proj_gemm<<<gProj, blk, 0, stream>>>(value, Wv, bv, Vh);

  dim3 gScore(SEQ / 128, SEQ / 128, BHCNT);         // (16, 16, 64)
  mha_score_gemm<<<gScore, blk, 0, stream>>>(Qh, Kh, attn);

  mha_row_stats<<<dim3(BHCNT * SEQ), blk, 0, stream>>>(attn, rowmax, rowrinv);

  dim3 gPV(SEQ / 128, BHCNT);                       // (16, 64)
  mha_pv_gemm<<<gPV, blk, 0, stream>>>(attn, rowmax, rowrinv, Vh, ctx);

  mha_out_gemm<<<gProj, blk, 0, stream>>>(ctx, Wo, bo, out);
}
